// TorchLMHeadCE_22101901705770
// MI455X (gfx1250) — matrix operand order repaired and checked by reading, not compiled, not run on hardware
//
#include <hip/hip_runtime.h>
#include <hip/hip_bf16.h>

// ---------------------------------------------------------------------------
// MI455X (gfx1250, wave32) fused LM-head JSD distillation loss.
//   logits = X @ W^T + b   (bf16 WMMA, f32 accumulate, double-buffered LDS)
//   log-softmax per row, generalized JSD (beta = 0.5), batch-mean.
// ---------------------------------------------------------------------------

typedef __attribute__((ext_vector_type(16))) __bf16 v16bf;
typedef __attribute__((ext_vector_type(8)))  __bf16 v8bf;
typedef __attribute__((ext_vector_type(8)))  float  v8f;

#define WAVE 32           // gfx1250 is wave32-only
#define TBLK 256          // 8 waves per workgroup
#define MT   128          // M tile
#define NT   128          // N tile
#define KT   64           // K chunk staged in LDS
#define LDS_STRIDE 72     // padded row stride (bf16 elems) -> 144B, kills bank conflicts

#define H_DIM 4096
#define BETA  0.5f

// ---------------------------------------------------------------------------
// Kernel 1: C[m0..m0+127, n0..n0+127] = X @ W^T + bias   (row-major f32 out)
// X: [M, K] f32 row-major.  W: [N, K] f32 row-major (LM head weight).
// Software-pipelined: global loads of chunk i+1 are issued before the WMMA
// burst of chunk i; LDS is double-buffered -> one barrier per chunk.
// Grid: x = M tile (fast) so X stays L2-resident and W streams exactly once.
// ---------------------------------------------------------------------------
__global__ void __launch_bounds__(TBLK)
gemm_logits_kernel(const float* __restrict__ X,
                   const float* __restrict__ W,
                   const float* __restrict__ bias,
                   float* __restrict__ out,
                   int K, int N)
{
    __shared__ __attribute__((aligned(16))) __bf16 Atile[2][MT][LDS_STRIDE];
    __shared__ __attribute__((aligned(16))) __bf16 Btile[2][NT][LDS_STRIDE];

    const int m0   = blockIdx.x * MT;    // M fast-varying: X tiles L2-resident
    const int n0   = blockIdx.y * NT;
    const int tid  = threadIdx.x;
    const int lane = tid & (WAVE - 1);
    const int wv   = tid >> 5;        // 0..7
    const int wm   = wv >> 1;         // 0..3 : wave row   (32 rows each)
    const int wn   = wv & 1;          // 0..1 : wave col   (64 cols each)
    const int hsel = lane >> 4;       // 0 or 1 (lane half)
    const int l15  = lane & 15;

    v8f acc[2][4] = {};               // 2 M-subtiles x 4 N-subtiles of 16x16 f32

    // LDS fill assignment: thread t covers one half-row (32 f32) of A and B.
    const int fr = tid >> 1;                 // row 0..127
    const int fk = (tid & 1) * 32;           // k half: 0 or 32
    const float* __restrict__ baseA = X + (size_t)(m0 + fr) * K + fk;
    const float* __restrict__ baseB = W + (size_t)(n0 + fr) * K + fk;

    float4 fa[8], fb[8];              // in-flight global stage (f32)

    // ---- prologue: fetch + stage chunk 0 ------------------------------------
    #pragma unroll
    for (int i = 0; i < 8; ++i) {
        fa[i] = *(const float4*)(baseA + 4 * i);
        fb[i] = *(const float4*)(baseB + 4 * i);
    }
    {
        __bf16 ta[32], tb[32];
        #pragma unroll
        for (int i = 0; i < 8; ++i) {
            ta[4*i+0] = (__bf16)fa[i].x; ta[4*i+1] = (__bf16)fa[i].y;
            ta[4*i+2] = (__bf16)fa[i].z; ta[4*i+3] = (__bf16)fa[i].w;
            tb[4*i+0] = (__bf16)fb[i].x; tb[4*i+1] = (__bf16)fb[i].y;
            tb[4*i+2] = (__bf16)fb[i].z; tb[4*i+3] = (__bf16)fb[i].w;
        }
        #pragma unroll
        for (int i = 0; i < 32; i += 8) {
            *(v8bf*)&Atile[0][fr][fk + i] = *(const v8bf*)&ta[i];
            *(v8bf*)&Btile[0][fr][fk + i] = *(const v8bf*)&tb[i];
        }
    }
    __syncthreads();

    int buf = 0;
    for (int kc = 0; kc < K; kc += KT) {
        const bool more = (kc + KT) < K;

        // ---- issue next chunk's global loads (latency hidden by WMMAs) -----
        if (more) {
            const float* __restrict__ srcA = baseA + kc + KT;
            const float* __restrict__ srcB = baseB + kc + KT;
            #pragma unroll
            for (int i = 0; i < 8; ++i) {
                fa[i] = *(const float4*)(srcA + 4 * i);
                fb[i] = *(const float4*)(srcB + 4 * i);
            }
        }

        // ---- two K=32 WMMA steps from LDS buffer `buf` ----------------------
        #pragma unroll
        for (int ks = 0; ks < 2; ++ks) {
            const int kb = ks * 32;

            // A fragments (16x32 bf16): lanes 0-15 -> K 0-7 & 16-23 of kb,
            //                           lanes 16-31 -> K 8-15 & 24-31.
            v16bf afrag[2];
            #pragma unroll
            for (int i = 0; i < 2; ++i) {
                const int row = wm * 32 + i * 16 + l15;
                const int ka  = kb + hsel * 8;
                v8bf lo = *(const v8bf*)&Atile[buf][row][ka];
                v8bf hi = *(const v8bf*)&Atile[buf][row][ka + 16];
                v16bf a;
                #pragma unroll
                for (int e = 0; e < 8; ++e) { a[e] = lo[e]; a[8 + e] = hi[e]; }
                afrag[i] = a;
            }

            // B fragments (32x16 bf16): lane holds column l15; lanes 0-15 hold
            // K 0-7 & 16-23, lanes 16-31 hold K 8-15 & 24-31.  Btile is [n][k] so
            // each group of 8 K-values is contiguous in LDS.
            v16bf bfrag[4];
            #pragma unroll
            for (int j = 0; j < 4; ++j) {
                const int col = wn * 64 + j * 16 + l15;
                const int kk  = kb + hsel * 8;
                v8bf lo = *(const v8bf*)&Btile[buf][col][kk];
                v8bf hi = *(const v8bf*)&Btile[buf][col][kk + 16];
                v16bf b;
                #pragma unroll
                for (int e = 0; e < 8; ++e) { b[e] = lo[e]; b[8 + e] = hi[e]; }
                bfrag[j] = b;
            }

            #pragma unroll
            for (int i = 0; i < 2; ++i)
                #pragma unroll
                for (int j = 0; j < 4; ++j)
                    acc[i][j] = __builtin_amdgcn_wmma_f32_16x16x32_bf16(
                        false, afrag[i], false, bfrag[j],
                        (short)0, acc[i][j], false, false);
        }

        // ---- convert + store next chunk into the other LDS buffer ----------
        if (more) {
            __bf16 ta[32], tb[32];
            #pragma unroll
            for (int i = 0; i < 8; ++i) {
                ta[4*i+0] = (__bf16)fa[i].x; ta[4*i+1] = (__bf16)fa[i].y;
                ta[4*i+2] = (__bf16)fa[i].z; ta[4*i+3] = (__bf16)fa[i].w;
                tb[4*i+0] = (__bf16)fb[i].x; tb[4*i+1] = (__bf16)fb[i].y;
                tb[4*i+2] = (__bf16)fb[i].z; tb[4*i+3] = (__bf16)fb[i].w;
            }
            const int nb = buf ^ 1;
            #pragma unroll
            for (int i = 0; i < 32; i += 8) {
                *(v8bf*)&Atile[nb][fr][fk + i] = *(const v8bf*)&ta[i];
                *(v8bf*)&Btile[nb][fr][fk + i] = *(const v8bf*)&tb[i];
            }
            __syncthreads();          // one barrier per K chunk
            buf = nb;
        }
    }

    // ---- epilogue: C layout VGPR r -> M=r (lanes 0-15) / M=r+8 (lanes 16-31)
    #pragma unroll
    for (int i = 0; i < 2; ++i) {
        const int rowbase = m0 + wm * 32 + i * 16 + hsel * 8;
        #pragma unroll
        for (int j = 0; j < 4; ++j) {
            const int col = n0 + wn * 64 + j * 16 + l15;
            const float bv = bias[col];
            #pragma unroll
            for (int r = 0; r < 8; ++r)
                out[(size_t)(rowbase + r) * N + col] = acc[i][j][r] + bv;
        }
    }
}

// ---------------------------------------------------------------------------
// Block-wide sum reduction (wave32-aware).
// ---------------------------------------------------------------------------
__device__ inline float block_sum(float v)
{
    #pragma unroll
    for (int off = 16; off > 0; off >>= 1)
        v += __shfl_xor(v, off, WAVE);
    __shared__ float tmp[TBLK / WAVE];
    const int lane = threadIdx.x & (WAVE - 1);
    const int w    = threadIdx.x >> 5;
    if (lane == 0) tmp[w] = v;
    __syncthreads();
    if (w == 0) {
        v = (lane < TBLK / WAVE) ? tmp[lane] : 0.0f;
        #pragma unroll
        for (int off = (TBLK / WAVE) >> 1; off > 0; off >>= 1)
            v += __shfl_xor(v, off, WAVE);
    }
    return v;   // valid on thread 0
}

// ---------------------------------------------------------------------------
// Kernel 2: per-row log-sum-exp (online), one block per (row, tensor).
// ---------------------------------------------------------------------------
__global__ void __launch_bounds__(TBLK)
row_lse_kernel(const float* __restrict__ logit_s,
               const float* __restrict__ logit_t,
               float* __restrict__ lse, int V)
{
    const int row = blockIdx.x;
    const float* __restrict__ p =
        ((blockIdx.y == 0) ? logit_s : logit_t) + (size_t)row * V;

    float m = -1e30f, s = 0.0f;
    for (int i = threadIdx.x; i < V; i += TBLK) {
        const float x = p[i];
        const float nm = fmaxf(m, x);
        s = s * __expf(m - nm) + __expf(x - nm);
        m = nm;
    }
    // pairwise (m,s) combine across the wave
    #pragma unroll
    for (int off = 16; off > 0; off >>= 1) {
        const float om = __shfl_xor(m, off, WAVE);
        const float os = __shfl_xor(s, off, WAVE);
        const float nm = fmaxf(m, om);
        s = s * __expf(m - nm) + os * __expf(om - nm);
        m = nm;
    }
    __shared__ float sm[TBLK / WAVE], ss[TBLK / WAVE];
    const int lane = threadIdx.x & (WAVE - 1);
    const int w    = threadIdx.x >> 5;
    if (lane == 0) { sm[w] = m; ss[w] = s; }
    __syncthreads();
    if (threadIdx.x == 0) {
        float M = sm[0], S = ss[0];
        #pragma unroll
        for (int k = 1; k < TBLK / WAVE; ++k) {
            const float nm = fmaxf(M, sm[k]);
            S = S * __expf(M - nm) + ss[k] * __expf(sm[k] - nm);
            M = nm;
        }
        lse[(size_t)blockIdx.y * gridDim.x + row] = M + __logf(S);
    }
}

// ---------------------------------------------------------------------------
// Kernel 3: per-row generalized JSD sum (beta = 0.5).
// ---------------------------------------------------------------------------
__global__ void __launch_bounds__(TBLK)
jsd_row_kernel(const float* __restrict__ logit_s,
               const float* __restrict__ logit_t,
               const float* __restrict__ lse,
               float* __restrict__ per_row, int V, int nrows)
{
    const int row = blockIdx.x;
    const float* __restrict__ qs = logit_s + (size_t)row * V;
    const float* __restrict__ ps = logit_t + (size_t)row * V;
    const float lq_off = lse[row];            // student LSE
    const float lp_off = lse[nrows + row];    // teacher LSE

    float acc = 0.0f;
    for (int i = threadIdx.x; i < V; i += TBLK) {
        const float lq = qs[i] - lq_off;      // log q
        const float lp = ps[i] - lp_off;      // log p
        const float q  = __expf(lq);
        const float p  = __expf(lp);
        const float mm = BETA * p + (1.0f - BETA) * q;
        const float lm = __logf(mm);
        acc += BETA * p * (lp - lm) + (1.0f - BETA) * q * (lq - lm);
    }
    const float total = block_sum(acc);
    if (threadIdx.x == 0) per_row[row] = total;
}

// ---------------------------------------------------------------------------
// Kernel 4: batch-mean of per-row values -> d_out[0].
// ---------------------------------------------------------------------------
__global__ void __launch_bounds__(TBLK)
mean_kernel(const float* __restrict__ per_row, float* __restrict__ out, int n)
{
    float acc = 0.0f;
    for (int i = threadIdx.x; i < n; i += TBLK) acc += per_row[i];
    const float total = block_sum(acc);
    if (threadIdx.x == 0) out[0] = total / (float)n;
}

// ---------------------------------------------------------------------------
extern "C" void kernel_launch(void* const* d_in, const int* in_sizes, int n_in,
                              void* d_out, int out_size, void* d_ws, size_t ws_size,
                              hipStream_t stream)
{
    const float* Xs = (const float*)d_in[0];   // student_input [M, H]
    const float* Xt = (const float*)d_in[1];   // teacher_input [M, H]
    const float* Ws = (const float*)d_in[2];   // W_student     [V, H]
    const float* bs = (const float*)d_in[3];   // b_student     [V]
    const float* Wt = (const float*)d_in[4];   // W_teacher     [V, H]
    const float* bt = (const float*)d_in[5];   // b_teacher     [V]

    const int K = H_DIM;
    const int M = in_sizes[0] / K;             // 2048
    const int V = in_sizes[3];                 // 32000

    float* ws = (float*)d_ws;
    const size_t NV = (size_t)M * V;
    float* logit_s = ws;
    float* logit_t = ws + NV;
    float* lse     = logit_t + NV;             // 2*M floats
    float* per_row = lse + 2 * (size_t)M;      // M floats

    const dim3 blk(TBLK);
    const dim3 gg(M / MT, V / NT);             // M fast -> X L2-resident, W streamed once
    gemm_logits_kernel<<<gg, blk, 0, stream>>>(Xs, Ws, bs, logit_s, K, V);
    gemm_logits_kernel<<<gg, blk, 0, stream>>>(Xt, Wt, bt, logit_t, K, V);

    row_lse_kernel<<<dim3(M, 2), blk, 0, stream>>>(logit_s, logit_t, lse, V);
    jsd_row_kernel<<<dim3(M), blk, 0, stream>>>(logit_s, logit_t, lse, per_row, V, M);
    mean_kernel<<<1, blk, 0, stream>>>(per_row, (float*)d_out, M);
}
